// DenoiseSum_28089086115869
// MI455X (gfx1250) — matrix operand order repaired and checked by reading, not compiled, not run on hardware
//
#include <hip/hip_runtime.h>
#include <math.h>

// ---------------------------------------------------------------------------
// Model dims (compile-time)
#define B_   8
#define D_   8
#define T_   64
#define O_   64
#define N_   64      // B*D
#define H_   512
#define HH_  256
#define WD_  256
#define V_   32000
#define DT_  512     // D*T
#define DS_  8

typedef __bf16 bf16_t;
typedef __attribute__((ext_vector_type(8)))  __bf16 v8bf;
typedef __attribute__((ext_vector_type(16))) __bf16 v16bf;
typedef __attribute__((ext_vector_type(8)))  float  v8f;

// ---------------------------------------------------------------------------
// Helpers
__device__ __forceinline__ float sigm(float x) { return 1.0f / (1.0f + __expf(-x)); }

__device__ __forceinline__ float wsum(float v) {
#pragma unroll
  for (int o = 16; o; o >>= 1) v += __shfl_xor(v, o, 32);
  return v;
}
__device__ __forceinline__ float wmaxf(float v) {
#pragma unroll
  for (int o = 16; o; o >>= 1) v = fmaxf(v, __shfl_xor(v, o, 32));
  return v;
}
// block reductions (blockDim.x == 256 -> 8 waves, wave32)
__device__ __forceinline__ float bsum(float v, float* sh) {
  int lane = threadIdx.x & 31, w = threadIdx.x >> 5, nw = blockDim.x >> 5;
  v = wsum(v);
  if (lane == 0) sh[w] = v;
  __syncthreads();
  if (threadIdx.x == 0) { float r = 0.f; for (int i = 0; i < nw; ++i) r += sh[i]; sh[0] = r; }
  __syncthreads();
  float r = sh[0];
  __syncthreads();
  return r;
}
__device__ __forceinline__ float bmax(float v, float* sh) {
  int lane = threadIdx.x & 31, w = threadIdx.x >> 5, nw = blockDim.x >> 5;
  v = wmaxf(v);
  if (lane == 0) sh[w] = v;
  __syncthreads();
  if (threadIdx.x == 0) { float r = -1e30f; for (int i = 0; i < nw; ++i) r = fmaxf(r, sh[i]); sh[0] = r; }
  __syncthreads();
  float r = sh[0];
  __syncthreads();
  return r;
}

__device__ __forceinline__ v16bf bf_load_2x8(const bf16_t* p0, const bf16_t* p1) {
  v8bf a = *(const v8bf*)p0;
  v8bf b = *(const v8bf*)p1;
  return __builtin_shufflevector(a, b, 0,1,2,3,4,5,6,7,8,9,10,11,12,13,14,15);
}

// ---------------------------------------------------------------------------
// WMMA GEMM (NT):  C[M,N] = A[M,K](bf16,row-major) x W[N,K](bf16,row-major)^T
//                  + bias[N] (opt) + Cin[M,N] (opt), act: 0=none 1=tanh 2=sigmoid
// 8 waves/block, each wave owns one 16x16 tile; K in steps of 64 via two
// independent v_wmma_f32_16x16x32_bf16 accumulator chains (hides the
// WMMA->WMMA RAW hazard NOPs and keeps loads in flight).
// Out-of-range rows are CLAMPED (not predicated): garbage only lands in
// accumulator rows/cols the epilogue never stores, and the inner loop stays
// branch-free (no exec save/restore per iteration).
// Fragment layouts per CDNA5 ISA 7.12.2:
//  A (16x32, 16-bit): lane<16 row=lane  holds K {kk..kk+7, kk+16..kk+23}
//                     lane>=16 row=lane-16 holds K {kk+8..kk+15, kk+24..kk+31}
//  B (32x16): lane<16 col=lane holds K {kk..kk+7, kk+16..kk+23}; lane>=16 col=lane-16 K {kk+8..kk+15, kk+24..kk+31}
//  C (f32, 8 VGPRs): vgpr r -> (row = r + (lane>=16?8:0), col = lane&15)
__global__ __launch_bounds__(256) void k_gemm_nt(
    const bf16_t* __restrict__ A, const bf16_t* __restrict__ W,
    const float* __restrict__ bias, const float* __restrict__ Cin,
    float* __restrict__ C, int M, int N, int K, int act)
{
  const int lane = threadIdx.x & 31;
  const int wave = threadIdx.x >> 5;
  const int n0 = (blockIdx.x * 8 + wave) * 16;
  const int m0 = blockIdx.y * 16;
  const bool hi = lane >= 16;
  const int l16 = lane & 15;

  const int arow = min(m0 + l16, M - 1);   // clamp: always a valid row
  const int wrow = min(n0 + l16, N - 1);
  const bf16_t* Ap = A + (size_t)arow * K;
  const bf16_t* Wp = W + (size_t)wrow * K;
  const int aoff = hi ? 8 : 0;
  const int woff = hi ? 8 : 0;

  v8f acc0, acc1;
#pragma unroll
  for (int i = 0; i < 8; ++i) { acc0[i] = 0.0f; acc1[i] = 0.0f; }

  int kk = 0;
  for (; kk + 64 <= K; kk += 64) {
    v16bf a0 = bf_load_2x8(Ap + kk + aoff,      Ap + kk + 16 + aoff);
    v16bf w0 = bf_load_2x8(Wp + kk + woff,      Wp + kk + woff + 16);
    v16bf a1 = bf_load_2x8(Ap + kk + 32 + aoff, Ap + kk + 48 + aoff);
    v16bf w1 = bf_load_2x8(Wp + kk + 32 + woff, Wp + kk + 48 + woff);
    if (kk + 128 < K) {                     // -> global_prefetch_b8
      __builtin_prefetch(Wp + kk + 128, 0, 3);
      __builtin_prefetch(Ap + kk + 128, 0, 3);
    }
    acc0 = __builtin_amdgcn_wmma_f32_16x16x32_bf16(false, a0, false, w0,
                                                   (short)0, acc0, false, false);
    acc1 = __builtin_amdgcn_wmma_f32_16x16x32_bf16(false, a1, false, w1,
                                                   (short)0, acc1, false, false);
  }
  for (; kk < K; kk += 32) {                // tail (K % 64 == 32)
    v16bf a0 = bf_load_2x8(Ap + kk + aoff, Ap + kk + 16 + aoff);
    v16bf w0 = bf_load_2x8(Wp + kk + woff, Wp + kk + woff + 16);
    acc0 = __builtin_amdgcn_wmma_f32_16x16x32_bf16(false, a0, false, w0,
                                                   (short)0, acc0, false, false);
  }

  const int col = n0 + l16;
  if (col < N) {
    const float bv = bias ? bias[col] : 0.0f;
    const int rbase = m0 + (hi ? 8 : 0);
#pragma unroll
    for (int r = 0; r < 8; ++r) {
      const int row = rbase + r;
      if (row < M) {
        float v = acc0[r] + acc1[r] + bv;
        if (Cin) v += Cin[(size_t)row * N + col];
        if (act == 1) v = tanhf(v);
        else if (act == 2) v = sigm(v);
        C[(size_t)row * N + col] = v;
      }
    }
  }
}

// ---------------------------------------------------------------------------
// Small support kernels
__global__ void k_f32_to_bf16(const float* __restrict__ s, bf16_t* __restrict__ d, int n) {
  int i = blockIdx.x * blockDim.x + threadIdx.x;
  if (i < n) d[i] = (bf16_t)s[i];
}

// xs: out row = t*N + n, token id at ids[n*T + t]; emb f32 -> bf16
__global__ void k_gather_xs(const int* __restrict__ ids, const float* __restrict__ emb,
                            bf16_t* __restrict__ out) {
  int row = blockIdx.x;                 // 0 .. T*N-1
  int t = row / N_, n = row % N_;
  int id = ids[n * T_ + t];
  int j = threadIdx.x;                  // 0..255
  out[(size_t)row * WD_ + j] = (bf16_t)emb[(size_t)id * WD_ + j];
}

// ys: out row = o*B + b, id = outids[b*O + o]
__global__ void k_gather_ys(const int* __restrict__ ids, const float* __restrict__ emb,
                            bf16_t* __restrict__ out) {
  int row = blockIdx.x;                 // 0 .. O*B-1
  int o = row / B_, b = row % B_;
  int id = ids[b * O_ + o];
  int j = threadIdx.x;
  out[(size_t)row * WD_ + j] = (bf16_t)emb[(size_t)id * WD_ + j];
}

// enc mask m[t*N+n] = min(1, tam+tpm)
__global__ void k_enc_mask(const float* __restrict__ tam, const float* __restrict__ tpm,
                           float* __restrict__ m) {
  int idx = blockIdx.x * blockDim.x + threadIdx.x;
  if (idx >= T_ * N_) return;
  int t = idx / N_, n = idx % N_;
  m[idx] = fminf(1.0f, tam[n * T_ + t] + tpm[n * T_ + t]);
}

// LSTM pointwise: gates[rows,4*Hh] (i,f,g,o) ; h,c [rows,Hh] in/out
__global__ void k_lstm_point(const float* __restrict__ gates, float* __restrict__ h,
                             float* __restrict__ c, const float* __restrict__ mask,
                             bf16_t* __restrict__ hbf, float* __restrict__ hcopy, int rows) {
  int idx = blockIdx.x * blockDim.x + threadIdx.x;
  if (idx >= rows * HH_) return;
  int row = idx / HH_, j = idx % HH_;
  const float* g = gates + (size_t)row * 4 * HH_;
  float i_ = sigm(g[j]);
  float f_ = sigm(g[HH_ + j]);
  float gg = tanhf(g[2 * HH_ + j]);
  float o_ = sigm(g[3 * HH_ + j]);
  float c2 = f_ * c[idx] + i_ * gg;
  float h2 = o_ * tanhf(c2);
  if (mask) { float m = mask[row]; h2 = h2 * m + h[idx] * (1.0f - m); }
  c[idx] = c2;
  h[idx] = h2;
  if (hbf)   hbf[idx] = (bf16_t)h2;
  if (hcopy) hcopy[idx] = h2;
}

// hs[(n*T+t), 0:512] = concat(fw_hs[t*N+n], bw_hs[t*N+n]); + bf16 copy
__global__ void k_pack_hs(const float* __restrict__ fw, const float* __restrict__ bw,
                          float* __restrict__ hs, bf16_t* __restrict__ hsbf) {
  int idx = blockIdx.x * blockDim.x + threadIdx.x;
  if (idx >= N_ * T_ * H_) return;
  int row = idx / H_, j = idx % H_;
  int n = row / T_, t = row % T_;
  float v = (j < HH_) ? fw[((size_t)(t * N_ + n)) * HH_ + j]
                      : bw[((size_t)(t * N_ + n)) * HH_ + (j - HH_)];
  hs[idx] = v;
  hsbf[idx] = (bf16_t)v;
}

// ht[n,0:512] = concat(fw_hs[T-1,n], bw_hs[0,n])
__global__ void k_make_ht(const float* __restrict__ fw, const float* __restrict__ bw,
                          float* __restrict__ ht, bf16_t* __restrict__ htbf) {
  int idx = blockIdx.x * blockDim.x + threadIdx.x;
  if (idx >= N_ * H_) return;
  int n = idx / H_, j = idx % H_;
  float v = (j < HH_) ? fw[((size_t)((T_ - 1) * N_ + n)) * HH_ + j]
                      : bw[(size_t)n * HH_ + (j - HH_)];
  ht[idx] = v;
  htbf[idx] = (bf16_t)v;
}

__global__ void k_mean(const float* __restrict__ ht, const float* __restrict__ dmask,
                       float* __restrict__ mean, bf16_t* __restrict__ meanbf) {
  int idx = blockIdx.x * blockDim.x + threadIdx.x;
  if (idx >= B_ * H_) return;
  int b = idx / H_, j = idx % H_;
  float s = 0.f, dm = 0.f;
  for (int d = 0; d < D_; ++d) { s += ht[(size_t)(b * D_ + d) * H_ + j]; dm += dmask[b * D_ + d]; }
  float v = s / dm;
  mean[idx] = v;
  meanbf[idx] = (bf16_t)v;
}

// h2 = ht + tanh(gk + gq[b])
__global__ void k_h2(const float* __restrict__ ht, const float* __restrict__ gk,
                     const float* __restrict__ gq, float* __restrict__ h2,
                     bf16_t* __restrict__ h2bf) {
  int idx = blockIdx.x * blockDim.x + threadIdx.x;
  if (idx >= N_ * H_) return;
  int n = idx / H_, j = idx % H_;
  int b = n / D_;
  float v = ht[idx] + tanhf(gk[idx] + gq[(size_t)b * H_ + j]);
  h2[idx] = v;
  h2bf[idx] = (bf16_t)v;
}

// per (b, channel): softmax over D of fk, mask, renorm, weighted sum of h2
__global__ void k_fuse_soft(const float* __restrict__ fk, const float* __restrict__ h2,
                            const float* __restrict__ dmask, float* __restrict__ s_out,
                            bf16_t* __restrict__ s_bf) {
  int idx = blockIdx.x * blockDim.x + threadIdx.x;
  if (idx >= B_ * H_) return;
  int b = idx / H_, j = idx % H_;
  float v[D_];
  float m = -1e30f;
  for (int d = 0; d < D_; ++d) { v[d] = fk[(size_t)(b * D_ + d) * H_ + j]; m = fmaxf(m, v[d]); }
  float s = 0.f;
  for (int d = 0; d < D_; ++d) { v[d] = __expf(v[d] - m); s += v[d]; }
  float norm = 0.f;
  for (int d = 0; d < D_; ++d) { v[d] = (v[d] / s) * dmask[b * D_ + d]; norm += v[d]; }
  float acc = 0.f;
  for (int d = 0; d < D_; ++d) acc += h2[(size_t)(b * D_ + d) * H_ + j] * (v[d] / norm);
  s_out[idx] = acc;
  s_bf[idx] = (bf16_t)acc;
}

// out_bf[b,0:512] = concat(st[b], st[b]) (disc input)
__global__ void k_dup(const float* __restrict__ st, bf16_t* __restrict__ out) {
  int idx = blockIdx.x * blockDim.x + threadIdx.x;
  if (idx >= B_ * H_) return;
  int b = idx / H_, j = idx % H_;
  out[idx] = (bf16_t)st[(size_t)b * HH_ + (j & (HH_ - 1))];
}

__global__ void k_disc(const float* __restrict__ zc, const float* __restrict__ cat,
                       float* __restrict__ acc) {
  int b = threadIdx.x;
  if (b >= B_) return;
  float m = -1e30f;
  for (int j = 0; j < DS_; ++j) m = fmaxf(m, zc[b * DS_ + j]);
  float s = 0.f;
  for (int j = 0; j < DS_; ++j) s += __expf(zc[b * DS_ + j] - m);
  float loss = 0.f;
  for (int j = 0; j < DS_; ++j) {
    float z = __expf(zc[b * DS_ + j] - m) / s;
    loss += -cat[b * DS_ + j] * __logf(z);
  }
  atomicAdd(acc, loss / (float)B_);
}

// attention scores: one wave per (b,pos)
__global__ void k_att_scores(const float* __restrict__ kt, const float* __restrict__ qt,
                             const float* __restrict__ cov, const float* __restrict__ covW,
                             const float* __restrict__ covB, const float* __restrict__ attW,
                             const float* __restrict__ attB, const int* __restrict__ cr,
                             float* __restrict__ e) {
  int gw = blockIdx.x * 8 + (threadIdx.x >> 5);   // 0..B*DT-1
  int lane = threadIdx.x & 31;
  int b = gw >> 9, pos = gw & (DT_ - 1);
  bool use_cov = (*cr != 0);
  float cv = cov[b * DT_ + pos];
  float p = 0.f;
  const float* krow = kt + ((size_t)(b * DT_ + pos)) * HH_;
  for (int j = lane; j < HH_; j += 32) {
    float q = qt[b * HH_ + j];
    if (use_cov) q += cv * covW[j] + covB[j];
    p += attW[j] * tanhf(krow[j] + q);
  }
  p = wsum(p);
  if (lane == 0) e[b * DT_ + pos] = p + attB[0];
}

// softmax over DT positions, mask+renorm, coverage update; one block per b
__global__ void k_att_softmax(const float* __restrict__ e, const float* __restrict__ tm,
                              float* __restrict__ at, float* __restrict__ cov,
                              float* __restrict__ cov_acc) {
  __shared__ float sh[8];
  int b = blockIdx.x, tid = threadIdx.x;
  float m = -1e30f;
  for (int i = tid; i < DT_; i += 256) m = fmaxf(m, e[b * DT_ + i]);
  m = bmax(m, sh);
  float s = 0.f;
  for (int i = tid; i < DT_; i += 256) s += __expf(e[b * DT_ + i] - m);
  s = bsum(s, sh);
  float norm = 0.f;
  for (int i = tid; i < DT_; i += 256) {
    float a = (__expf(e[b * DT_ + i] - m) / s) * tm[b * DT_ + i];
    at[b * DT_ + i] = a;
    norm += a;
  }
  norm = bsum(norm, sh);
  float covp = 0.f;
  for (int i = tid; i < DT_; i += 256) {
    float a = at[b * DT_ + i] / norm;
    at[b * DT_ + i] = a;
    float c = cov[b * DT_ + i];
    covp += fminf(c, a);
    cov[b * DT_ + i] = c + a;
  }
  covp = bsum(covp, sh);
  if (tid == 0) atomicAdd(cov_acc, covp);
}

// vt[b,:] = sum_pos at[b,pos]*hs[b,pos,:] ; one block per b
__global__ void k_wsum_ctx(const float* __restrict__ at, const float* __restrict__ hs,
                           float* __restrict__ vt, bf16_t* __restrict__ vtbf) {
  int b = blockIdx.x, j = threadIdx.x;   // j in 0..255, handles j and j+256
  float s0 = 0.f, s1 = 0.f;
  for (int p = 0; p < DT_; ++p) {
    float a = at[b * DT_ + p];
    const float* hp = hs + ((size_t)(b * DT_ + p)) * H_;
    s0 += a * hp[j];
    s1 += a * hp[j + HH_];
  }
  vt[(size_t)b * H_ + j] = s0;
  vt[(size_t)b * H_ + j + HH_] = s1;
  vtbf[(size_t)b * H_ + j] = (bf16_t)s0;
  vtbf[(size_t)b * H_ + j + HH_] = (bf16_t)s1;
}

// gate = sigmoid(W . [s(256), v(512)] + b) ; one wave per b (block of 256)
__global__ void k_gate(const float* __restrict__ s, const float* __restrict__ v,
                       const float* __restrict__ W, const float* __restrict__ bias,
                       float* __restrict__ out) {
  int b = threadIdx.x >> 5, lane = threadIdx.x & 31;
  float p = 0.f;
  for (int j = lane; j < HH_; j += 32) p += W[j] * s[b * HH_ + j];
  for (int j = lane; j < H_; j += 32) p += W[HH_ + j] * v[b * H_ + j];
  p = wsum(p);
  if (lane == 0) out[b] = sigm(p + bias[0]);
}

// NLL for one decoder step: one block per b
__global__ void k_nll(const float* __restrict__ la, const float* __restrict__ lp,
                      const float* __restrict__ ap, const int* __restrict__ inp,
                      const int* __restrict__ outids, const float* __restrict__ om,
                      const float* __restrict__ delta, const float* __restrict__ gamma,
                      int step, float* __restrict__ acc) {
  __shared__ float sh[8];
  int b = blockIdx.x, tid = threadIdx.x;
  int tgt = outids[b * O_ + step + 1];
  float omv = om[b * O_ + step + 1];
  float mA = -1e30f, mP = -1e30f;
  for (int i = tid; i < V_; i += 256) {
    mA = fmaxf(mA, la[(size_t)b * V_ + i]);
    mP = fmaxf(mP, lp[(size_t)b * V_ + i]);
  }
  mA = bmax(mA, sh);
  mP = bmax(mP, sh);
  float sA = 0.f, sP = 0.f;
  for (int i = tid; i < V_; i += 256) {
    sA += __expf(la[(size_t)b * V_ + i] - mA);
    sP += __expf(lp[(size_t)b * V_ + i] - mP);
  }
  sA = bsum(sA, sh);
  sP = bsum(sP, sh);
  float pc = 0.f;
  for (int i = tid; i < DT_; i += 256)
    if (inp[b * DT_ + i] == tgt) pc += ap[b * DT_ + i];
  pc = bsum(pc, sh);
  if (tid == 0) {
    float pw = __expf(la[(size_t)b * V_ + tgt] - mA) / sA;
    float pg = __expf(lp[(size_t)b * V_ + tgt] - mP) / sP;
    float d = delta[b], g = gamma[b];
    float pt = d * pw + (1.0f - d) * (g * pg + (1.0f - g) * pc);
    atomicAdd(acc, -__logf(pt + 1e-9f) * omv / (float)(B_ * (O_ - 1)));
  }
}

__global__ void k_finalize(const float* gen, const float* disc, const float* cov,
                           const int* cr, float* out) {
  out[0] = gen[0] + disc[0] + (float)(*cr) * (cov[0] / (float)(B_ * (O_ - 1)));
}

// ---------------------------------------------------------------------------
// Host orchestration
static inline void gemm(hipStream_t s, const bf16_t* A, const bf16_t* W, const float* bias,
                        const float* Cin, float* C, int M, int N, int K, int act) {
  dim3 g((N + 127) / 128, (M + 15) / 16);
  k_gemm_nt<<<g, 256, 0, s>>>(A, W, bias, Cin, C, M, N, K, act);
}

// input indices (setup_inputs flattening order)
enum {
  IN_INPUT = 0, IN_DAM, IN_DPM, IN_TAM, IN_TPM, IN_OUT, IN_OMASK, IN_CAT, IN_CR,
  P_EMB,
  P_FW_WIH, P_FW_WHH, P_FW_B, P_BW_WIH, P_BW_WHH, P_BW_B,
  P_DAK_W, P_DAK_B, P_DAQ_W, P_DAQ_B, P_DPK_W, P_DPK_B, P_DPQ_W, P_DPQ_B,
  P_FAK_W, P_FAK_B, P_FPK_W, P_FPK_B, P_FAT_W, P_FAT_B, P_FPT_W, P_FPT_B,
  P_DSI_W, P_DSI_B, P_DSC_W, P_DSC_B,
  P_AD_WIH, P_AD_WHH, P_AD_B, P_PD_WIH, P_PD_WHH, P_PD_B,
  P_AK_W, P_AK_B, P_AQ_W, P_AQ_B, P_AW_W, P_AW_B,
  P_ACTX_W, P_ACTX_B, P_AHID_W, P_AHID_B, P_ACOV_W, P_ACOV_B,
  P_PK_W, P_PK_B, P_PQ_W, P_PQ_B, P_PW_W, P_PW_B,
  P_PCTX_W, P_PCTX_B, P_PHID_W, P_PHID_B,
  P_PG_W, P_PG_B, P_PCOV_W, P_PCOV_B, P_FG_W, P_FG_B
};

extern "C" void kernel_launch(void* const* d_in, const int* in_sizes, int n_in,
                              void* d_out, int out_size, void* d_ws, size_t ws_size,
                              hipStream_t stream) {
  (void)in_sizes; (void)n_in; (void)out_size; (void)ws_size;
  auto fin = [&](int i) { return (const float*)d_in[i]; };
  auto iin = [&](int i) { return (const int*)d_in[i]; };

  // ---- workspace arena ----
  size_t off = 0;
  auto alloc = [&](size_t bytes) {
    void* r = (char*)d_ws + off;
    off = (off + bytes + 255) & ~(size_t)255;
    return r;
  };
  auto allocF  = [&](size_t n) { return (float*)alloc(n * sizeof(float)); };
  auto allocB  = [&](size_t n) { return (bf16_t*)alloc(n * sizeof(bf16_t)); };

  auto cvt = [&](int idx, bf16_t* dst, size_t n) {
    k_f32_to_bf16<<<(int)((n + 255) / 256), 256, 0, stream>>>(fin(idx), dst, (int)n);
  };

  // bf16 weight copies (L2-resident working set; vocab mats ~98MB total in bf16)
  bf16_t* fw_wih = allocB(1024 * 256); bf16_t* fw_whh = allocB(1024 * 256);
  bf16_t* bw_wih = allocB(1024 * 256); bf16_t* bw_whh = allocB(1024 * 256);
  bf16_t* ad_wih = allocB(1024 * 256); bf16_t* ad_whh = allocB(1024 * 256);
  bf16_t* pd_wih = allocB(1024 * 256); bf16_t* pd_whh = allocB(1024 * 256);
  bf16_t* dak = allocB(512 * 512); bf16_t* daq = allocB(512 * 512);
  bf16_t* dpk = allocB(512 * 512); bf16_t* dpq = allocB(512 * 512);
  bf16_t* fak = allocB(512 * 512); bf16_t* fpk = allocB(512 * 512);
  bf16_t* fat = allocB(256 * 512); bf16_t* fpt = allocB(256 * 512);
  bf16_t* dsi = allocB(512 * 512); bf16_t* dsc = allocB(8 * 512);
  bf16_t* akw = allocB(256 * 512); bf16_t* pkw = allocB(256 * 512);
  bf16_t* aqw = allocB(256 * 256); bf16_t* pqw = allocB(256 * 256);
  bf16_t* actx = allocB((size_t)V_ * 512); bf16_t* ahid = allocB((size_t)V_ * 256);
  bf16_t* pctx = allocB((size_t)V_ * 512); bf16_t* phid = allocB((size_t)V_ * 256);

  cvt(P_FW_WIH, fw_wih, 1024 * 256); cvt(P_FW_WHH, fw_whh, 1024 * 256);
  cvt(P_BW_WIH, bw_wih, 1024 * 256); cvt(P_BW_WHH, bw_whh, 1024 * 256);
  cvt(P_AD_WIH, ad_wih, 1024 * 256); cvt(P_AD_WHH, ad_whh, 1024 * 256);
  cvt(P_PD_WIH, pd_wih, 1024 * 256); cvt(P_PD_WHH, pd_whh, 1024 * 256);
  cvt(P_DAK_W, dak, 512 * 512); cvt(P_DAQ_W, daq, 512 * 512);
  cvt(P_DPK_W, dpk, 512 * 512); cvt(P_DPQ_W, dpq, 512 * 512);
  cvt(P_FAK_W, fak, 512 * 512); cvt(P_FPK_W, fpk, 512 * 512);
  cvt(P_FAT_W, fat, 256 * 512); cvt(P_FPT_W, fpt, 256 * 512);
  cvt(P_DSI_W, dsi, 512 * 512); cvt(P_DSC_W, dsc, 8 * 512);
  cvt(P_AK_W, akw, 256 * 512); cvt(P_PK_W, pkw, 256 * 512);
  cvt(P_AQ_W, aqw, 256 * 256); cvt(P_PQ_W, pqw, 256 * 256);
  cvt(P_ACTX_W, actx, (size_t)V_ * 512); cvt(P_AHID_W, ahid, (size_t)V_ * 256);
  cvt(P_PCTX_W, pctx, (size_t)V_ * 512); cvt(P_PHID_W, phid, (size_t)V_ * 256);

  // activations
  bf16_t* xs_bf = allocB((size_t)T_ * N_ * WD_);
  bf16_t* ys_bf = allocB((size_t)O_ * B_ * WD_);
  float* encmask = allocF((size_t)T_ * N_);
  float* Gx_fw = allocF((size_t)T_ * N_ * 1024);
  float* Gx_bw = allocF((size_t)T_ * N_ * 1024);
  float* gates_e = allocF((size_t)N_ * 1024);
  float* h_fw = allocF(N_ * HH_); float* c_fw = allocF(N_ * HH_);
  float* h_bw = allocF(N_ * HH_); float* c_bw = allocF(N_ * HH_);
  bf16_t* hfw_bf = allocB(N_ * HH_); bf16_t* hbw_bf = allocB(N_ * HH_);
  float* fw_hs = allocF((size_t)T_ * N_ * HH_);
  float* bw_hs = allocF((size_t)T_ * N_ * HH_);
  float* hs = allocF((size_t)N_ * T_ * H_);
  bf16_t* hs_bf = allocB((size_t)N_ * T_ * H_);
  float* ht = allocF(N_ * H_); bf16_t* ht_bf = allocB(N_ * H_);
  float* kt_att = allocF((size_t)N_ * T_ * HH_);
  float* kt_pnt = allocF((size_t)N_ * T_ * HH_);
  float* mean = allocF(B_ * H_); bf16_t* mean_bf = allocB(B_ * H_);
  float* gk = allocF(N_ * H_); float* gq = allocF(B_ * H_);
  float* h2 = allocF(N_ * H_); bf16_t* h2_bf = allocB(N_ * H_);
  float* fk = allocF(N_ * H_);
  float* s_f = allocF(B_ * H_); bf16_t* s_bf = allocB(B_ * H_);
  float* st_att0 = allocF(B_ * HH_); float* st_pnt0 = allocF(B_ * HH_);
  bf16_t* stcat_bf = allocB(B_ * H_);
  float* zi = allocF(B_ * H_); bf16_t* zi_bf = allocB(B_ * H_);
  float* zc = allocF(B_ * DS_);
  float* sa = allocF(B_ * HH_); float* ca = allocF(B_ * HH_);
  float* sp = allocF(B_ * HH_); float* cp = allocF(B_ * HH_);
  bf16_t* sa_bf = allocB(B_ * HH_); bf16_t* sp_bf = allocB(B_ * HH_);
  float* gates_a = allocF(B_ * 1024); float* gates_p = allocF(B_ * 1024);
  float* qt_a = allocF(B_ * HH_); float* qt_p = allocF(B_ * HH_);
  float* e_a = allocF(B_ * DT_); float* e_p = allocF(B_ * DT_);
  float* at_a = allocF(B_ * DT_); float* at_p = allocF(B_ * DT_);
  float* cov_a = allocF(B_ * DT_); float* cov_p = allocF(B_ * DT_);
  float* vt_a = allocF(B_ * H_); float* vt_p = allocF(B_ * H_);
  bf16_t* vta_bf = allocB(B_ * H_); bf16_t* vtp_bf = allocB(B_ * H_);
  float* logits_a = allocF((size_t)B_ * V_);
  float* logits_p = allocF((size_t)B_ * V_);
  float* delta = allocF(B_); float* gamma = allocF(B_);
  float* accs = allocF(4);   // [gen, disc, cov]

  // ---- zero-init recurrent state & accumulators ----
  hipMemsetAsync(h_fw, 0, N_ * HH_ * 4, stream);
  hipMemsetAsync(c_fw, 0, N_ * HH_ * 4, stream);
  hipMemsetAsync(h_bw, 0, N_ * HH_ * 4, stream);
  hipMemsetAsync(c_bw, 0, N_ * HH_ * 4, stream);
  hipMemsetAsync(hfw_bf, 0, N_ * HH_ * 2, stream);
  hipMemsetAsync(hbw_bf, 0, N_ * HH_ * 2, stream);
  hipMemsetAsync(ca, 0, B_ * HH_ * 4, stream);
  hipMemsetAsync(cp, 0, B_ * HH_ * 4, stream);
  hipMemsetAsync(cov_a, 0, B_ * DT_ * 4, stream);
  hipMemsetAsync(cov_p, 0, B_ * DT_ * 4, stream);
  hipMemsetAsync(accs, 0, 4 * 4, stream);

  // ---- embedding gathers + encoder mask ----
  k_gather_xs<<<T_ * N_, 256, 0, stream>>>(iin(IN_INPUT), fin(P_EMB), xs_bf);
  k_gather_ys<<<O_ * B_, 256, 0, stream>>>(iin(IN_OUT), fin(P_EMB), ys_bf);
  k_enc_mask<<<(T_ * N_ + 255) / 256, 256, 0, stream>>>(fin(IN_TAM), fin(IN_TPM), encmask);

  // ---- encoder input projections (time-batched big GEMMs) ----
  gemm(stream, xs_bf, fw_wih, fin(P_FW_B), nullptr, Gx_fw, T_ * N_, 1024, 256, 0);
  gemm(stream, xs_bf, bw_wih, fin(P_BW_B), nullptr, Gx_bw, T_ * N_, 1024, 256, 0);

  // ---- forward recurrence ----
  for (int t = 0; t < T_; ++t) {
    gemm(stream, hfw_bf, fw_whh, nullptr, Gx_fw + (size_t)t * N_ * 1024, gates_e,
         N_, 1024, 256, 0);
    k_lstm_point<<<(N_ * HH_ + 255) / 256, 256, 0, stream>>>(
        gates_e, h_fw, c_fw, encmask + t * N_, hfw_bf, fw_hs + (size_t)t * N_ * HH_, N_);
  }
  // ---- backward recurrence ----
  for (int s = 0; s < T_; ++s) {
    int t = T_ - 1 - s;
    gemm(stream, hbw_bf, bw_whh, nullptr, Gx_bw + (size_t)t * N_ * 1024, gates_e,
         N_, 1024, 256, 0);
    k_lstm_point<<<(N_ * HH_ + 255) / 256, 256, 0, stream>>>(
        gates_e, h_bw, c_bw, encmask + t * N_, hbw_bf, bw_hs + (size_t)t * N_ * HH_, N_);
  }

  k_pack_hs<<<(N_ * T_ * H_ + 255) / 256, 256, 0, stream>>>(fw_hs, bw_hs, hs, hs_bf);
  k_make_ht<<<(N_ * H_ + 255) / 256, 256, 0, stream>>>(fw_hs, bw_hs, ht, ht_bf);

  // ---- attention keys ----
  gemm(stream, hs_bf, akw, fin(P_AK_B), nullptr, kt_att, N_ * T_, HH_, H_, 0);
  gemm(stream, hs_bf, pkw, fin(P_PK_B), nullptr, kt_pnt, N_ * T_, HH_, H_, 0);

  // ---- fuse (att / pnt) ----
  struct Fuse { const bf16_t *dk, *dq, *fk_w, *ft; int dkb, dqb, fkb, ftb, dm; float* st; };
  Fuse fz[2] = {
    { dak, daq, fak, fat, P_DAK_B, P_DAQ_B, P_FAK_B, P_FAT_B, IN_DAM, st_att0 },
    { dpk, dpq, fpk, fpt, P_DPK_B, P_DPQ_B, P_FPK_B, P_FPT_B, IN_DPM, st_pnt0 },
  };
  for (int i = 0; i < 2; ++i) {
    k_mean<<<(B_ * H_ + 255) / 256, 256, 0, stream>>>(ht, fin(fz[i].dm), mean, mean_bf);
    gemm(stream, ht_bf, fz[i].dk, fin(fz[i].dkb), nullptr, gk, N_, H_, H_, 0);
    gemm(stream, mean_bf, fz[i].dq, fin(fz[i].dqb), nullptr, gq, B_, H_, H_, 0);
    k_h2<<<(N_ * H_ + 255) / 256, 256, 0, stream>>>(ht, gk, gq, h2, h2_bf);
    gemm(stream, h2_bf, fz[i].fk_w, fin(fz[i].fkb), nullptr, fk, N_, H_, H_, 0);
    k_fuse_soft<<<(B_ * H_ + 255) / 256, 256, 0, stream>>>(fk, h2, fin(fz[i].dm), s_f, s_bf);
    gemm(stream, s_bf, fz[i].ft, fin(fz[i].ftb), nullptr, fz[i].st, B_, HH_, H_, 1);
  }

  // ---- discriminator loss ----
  k_dup<<<(B_ * H_ + 255) / 256, 256, 0, stream>>>(st_pnt0, stcat_bf);
  gemm(stream, stcat_bf, dsi, fin(P_DSI_B), nullptr, zi, B_, H_, H_, 1);
  k_f32_to_bf16<<<(B_ * H_ + 255) / 256, 256, 0, stream>>>(zi, zi_bf, B_ * H_);
  gemm(stream, zi_bf, dsc, fin(P_DSC_B), nullptr, zc, B_, DS_, H_, 0);
  k_disc<<<1, 64, 0, stream>>>(zc, fin(IN_CAT), accs + 1);

  // ---- decoder init ----
  hipMemcpyAsync(sa, st_att0, B_ * HH_ * 4, hipMemcpyDeviceToDevice, stream);
  hipMemcpyAsync(sp, st_pnt0, B_ * HH_ * 4, hipMemcpyDeviceToDevice, stream);
  k_f32_to_bf16<<<(B_ * HH_ + 255) / 256, 256, 0, stream>>>(sa, sa_bf, B_ * HH_);
  k_f32_to_bf16<<<(B_ * HH_ + 255) / 256, 256, 0, stream>>>(sp, sp_bf, B_ * HH_);

  const int* cr = iin(IN_CR);

  // ---- decoder: 63 steps ----
  for (int t = 0; t < O_ - 1; ++t) {
    const bf16_t* yt = ys_bf + (size_t)t * B_ * WD_;

    // attention branch
    gemm(stream, yt, ad_wih, fin(P_AD_B), nullptr, gates_a, B_, 1024, 256, 0);
    gemm(stream, sa_bf, ad_whh, nullptr, gates_a, gates_a, B_, 1024, 256, 0);
    k_lstm_point<<<(B_ * HH_ + 255) / 256, 256, 0, stream>>>(
        gates_a, sa, ca, nullptr, sa_bf, nullptr, B_);
    gemm(stream, sa_bf, aqw, fin(P_AQ_B), nullptr, qt_a, B_, HH_, HH_, 0);
    k_att_scores<<<(B_ * DT_) / 8, 256, 0, stream>>>(
        kt_att, qt_a, cov_a, fin(P_ACOV_W), fin(P_ACOV_B), fin(P_AW_W), fin(P_AW_B), cr, e_a);
    k_att_softmax<<<B_, 256, 0, stream>>>(e_a, fin(IN_TAM), at_a, cov_a, accs + 2);
    k_wsum_ctx<<<B_, 256, 0, stream>>>(at_a, hs, vt_a, vta_bf);
    gemm(stream, vta_bf, actx, fin(P_ACTX_B), nullptr, logits_a, B_, V_, H_, 0);
    gemm(stream, sa_bf, ahid, fin(P_AHID_B), logits_a, logits_a, B_, V_, HH_, 0);
    k_gate<<<1, 256, 0, stream>>>(sa, vt_a, fin(P_FG_W), fin(P_FG_B), delta);

    // pointer branch
    gemm(stream, yt, pd_wih, fin(P_PD_B), nullptr, gates_p, B_, 1024, 256, 0);
    gemm(stream, sp_bf, pd_whh, nullptr, gates_p, gates_p, B_, 1024, 256, 0);
    k_lstm_point<<<(B_ * HH_ + 255) / 256, 256, 0, stream>>>(
        gates_p, sp, cp, nullptr, sp_bf, nullptr, B_);
    gemm(stream, sp_bf, pqw, fin(P_PQ_B), nullptr, qt_p, B_, HH_, HH_, 0);
    k_att_scores<<<(B_ * DT_) / 8, 256, 0, stream>>>(
        kt_pnt, qt_p, cov_p, fin(P_PCOV_W), fin(P_PCOV_B), fin(P_PW_W), fin(P_PW_B), cr, e_p);
    k_att_softmax<<<B_, 256, 0, stream>>>(e_p, fin(IN_TPM), at_p, cov_p, accs + 2);
    k_wsum_ctx<<<B_, 256, 0, stream>>>(at_p, hs, vt_p, vtp_bf);
    gemm(stream, vtp_bf, pctx, fin(P_PCTX_B), nullptr, logits_p, B_, V_, H_, 0);
    gemm(stream, sp_bf, phid, fin(P_PHID_B), logits_p, logits_p, B_, V_, HH_, 0);
    k_gate<<<1, 256, 0, stream>>>(sp, vt_p, fin(P_PG_W), fin(P_PG_B), gamma);

    // combine + NLL at target token only
    k_nll<<<B_, 256, 0, stream>>>(logits_a, logits_p, at_p, iin(IN_INPUT), iin(IN_OUT),
                                  fin(IN_OMASK), delta, gamma, t, accs + 0);
  }

  k_finalize<<<1, 1, 0, stream>>>(accs + 0, accs + 1, accs + 2, cr, (float*)d_out);
}
